// LSTMCell_72997264163163
// MI455X (gfx1250) — matrix operand order repaired and checked by reading, not compiled, not run on hardware
//
#include <hip/hip_runtime.h>
#include <hip/hip_bf16.h>
#include <stdint.h>

// LSTM cell for gfx1250 (MI455X).
// Pass 1: split fp32 -> (hi,lo) bf16 planes in workspace.
// Pass 2: 256x(4x32) fused GEMM (3-pass split-bf16 WMMA, 4x4 wave blocking)
//         + gate epilogue, double-buffered global_load_async_to_lds_b128.

typedef __attribute__((ext_vector_type(16))) __bf16 v16bf;
typedef __attribute__((ext_vector_type(8)))  __bf16 v8bf;
typedef __attribute__((ext_vector_type(8)))  float  v8f;

#define BQ   8192
#define HQ   1024
#define KQ   2048      // IN + H
#define BMF  128       // fallback: batch rows per workgroup
#define BMG  256       // async GEMM: batch rows per workgroup
#define BH   32        // h-columns per workgroup (x4 gates internally)
#define KT   32        // K per step (= WMMA bf16 K)
#define LDSS 40        // LDS row stride in bf16 elems (32 + 8 pad)

// LDS row-plane map for the async GEMM (per buffer, 768 rows x LDSS):
//   rows [  0,256) : A hi      rows [256,512) : A lo
//   rows [512,640) : W hi      rows [640,768) : W lo
#define ROWS_PB 768
#define BUFSTRIDE (ROWS_PB * LDSS * 2)   // bytes per LDS buffer = 61440

union FragU { v16bf v; v8bf h[2]; };

__device__ __forceinline__ unsigned short f2bf_rn(float x) {
  unsigned u = __builtin_bit_cast(unsigned, x);
  unsigned r = u + 0x7FFFu + ((u >> 16) & 1u);
  return (unsigned short)(r >> 16);
}
__device__ __forceinline__ float bfbits2f(unsigned short b) {
  return __builtin_bit_cast(float, ((unsigned)b) << 16);
}
__device__ __forceinline__ float sigmoidf_fast(float x) {
  return 1.0f / (1.0f + __expf(-x));
}
__device__ __forceinline__ float tanhf_fast(float x) {
  return 2.0f * sigmoidf_fast(2.0f * x) - 1.0f;
}

// ---------------------------------------------------------------------------
// Pass 1: split [rows x 2048] fp32 (cols 0-1023 from src0, 1024-2047 from
// src1, each row-major stride 1024) into hi/lo bf16 planes [rows x 2048].
// ---------------------------------------------------------------------------
__global__ void __launch_bounds__(256)
split_bf16_kernel(const float* __restrict__ src0, const float* __restrict__ src1,
                  __bf16* __restrict__ dhi, __bf16* __restrict__ dlo, int rows)
{
  int idx = blockIdx.x * 256 + threadIdx.x;   // one 4-elem chunk per thread
  int r   = idx >> 9;                         // 512 chunks per row
  int c   = (idx & 511) << 2;
  if (r >= rows) return;
  const float* s = (c < 1024) ? (src0 + (size_t)r * 1024 + c)
                              : (src1 + (size_t)r * 1024 + (c - 1024));
  const float4 x = *(const float4*)s;
  float v[4] = { x.x, x.y, x.z, x.w };
  unsigned long long hp = 0ull, lp = 0ull;
  #pragma unroll
  for (int j = 0; j < 4; ++j) {
    unsigned short hb = f2bf_rn(v[j]);
    unsigned short lb = f2bf_rn(v[j] - bfbits2f(hb));
    hp |= (unsigned long long)hb << (16 * j);
    lp |= (unsigned long long)lb << (16 * j);
  }
  *(unsigned long long*)(dhi + (size_t)r * KQ + c) = hp;
  *(unsigned long long*)(dlo + (size_t)r * KQ + c) = lp;
}

// ---------------------------------------------------------------------------
// Pass 2: GEMM + LSTM epilogue. 4x4 wave blocking, async-to-LDS double buffer.
// ws layout (bf16, contiguous): Xhi[8192x2048] Xlo Whi[4096x2048] Wlo.
// ---------------------------------------------------------------------------
__global__ void __launch_bounds__(256)
lstm_gemm_bf16(const __bf16* __restrict__ wsbase,   // = Xhi
               const float* __restrict__ state,
               const float* __restrict__ bx, const float* __restrict__ bh,
               float* __restrict__ out)
{
  __shared__ __bf16 lds[2][ROWS_PB][LDSS];    // 2 x 61440 B = 120 KiB

  const int tid    = threadIdx.x;
  const int lane   = tid & 31;
  const int wave   = tid >> 5;
  const int wave_m = wave >> 1;               // 0..3 : 64-row slab
  const int wave_n = wave & 1;                // 0..1 : 16-col half
  const int l16    = lane & 15;
  const int lhi    = lane >> 4;

  const int row0 = blockIdx.x * BMG;
  const int h0   = blockIdx.y * BH;

  const float* c_prev = state + (size_t)BQ * HQ;   // state[1]

  const size_t nX = (size_t)BQ * KQ;          // elems per X plane
  const size_t nW = (size_t)(4 * HQ) * KQ;    // elems per W plane

  // ---- per-thread async-copy plan: 3072 16B chunks / 256 threads = 12 ----
  unsigned goff[12];   // byte offset from wsbase (fits: 96 MiB < 2^31)
  unsigned loff[12];   // LDS byte offset within buffer 0
  #pragma unroll
  for (int i = 0; i < 12; ++i) {
    int cid = tid + i * 256;                  // 0..3071
    size_t gbyte; int ldsrow, cc;
    if (cid < 2048) {                         // A planes (hi:0..1023, lo:1024..)
      int pl = cid >> 10;
      int q  = cid & 1023;
      int r  = q >> 2;                        // 0..255
      cc     = (q & 3) * 8;
      gbyte  = ((pl ? nX : 0) + (size_t)(row0 + r) * KQ + cc) * 2;
      ldsrow = pl * 256 + r;
    } else {                                  // W planes
      int w  = cid - 2048;
      int pl = w >> 9;
      int q  = w & 511;
      int r  = q >> 2;                        // 0..127 = gate*32 + hl
      int gate = r >> 5, hl = r & 31;
      cc     = (q & 3) * 8;
      gbyte  = (2 * nX + (pl ? nW : 0) + (size_t)(gate * 1024 + h0 + hl) * KQ + cc) * 2;
      ldsrow = 512 + pl * 128 + r;
    }
    goff[i] = (unsigned)gbyte;
    loff[i] = (unsigned)(uintptr_t)&lds[0][ldsrow][cc];
  }

  #define ISSUE_TILE(PBUF, KKV)                                              \
    do {                                                                     \
      _Pragma("unroll")                                                      \
      for (int _i = 0; _i < 12; ++_i) {                                      \
        unsigned _vo = goff[_i] + (unsigned)((KKV) * 2);                     \
        unsigned _la = loff[_i] + (unsigned)(PBUF) * (unsigned)BUFSTRIDE;    \
        asm volatile("global_load_async_to_lds_b128 %0, %1, %2"              \
                     :: "v"(_la), "v"(_vo), "s"(wsbase) : "memory");         \
      }                                                                      \
    } while (0)

  v8f acc[4][4];
  #pragma unroll
  for (int mt = 0; mt < 4; ++mt)
    #pragma unroll
    for (int g = 0; g < 4; ++g)
      acc[mt][g] = (v8f){0.f,0.f,0.f,0.f,0.f,0.f,0.f,0.f};

  // prologue: fill buffer 0
  ISSUE_TILE(0, 0);
  asm volatile("s_wait_asynccnt 0x0" ::: "memory");
  __syncthreads();

  for (int kk = 0; kk < KQ; kk += KT) {
    const int p = (kk >> 5) & 1;
    if (kk + KT < KQ) ISSUE_TILE(p ^ 1, kk + KT);

    // ---- B (W) fragments, held across the mt loop ----
    FragU bfr[4][2];   // [gate][hi/lo]
    #pragma unroll
    for (int g = 0; g < 4; ++g) {
      int r  = g * 32 + wave_n * 16 + l16;
      int kb = lhi * 8;
      bfr[g][0].h[0] = *(const v8bf*)&lds[p][512 + r][kb];
      bfr[g][0].h[1] = *(const v8bf*)&lds[p][512 + r][kb + 16];
      bfr[g][1].h[0] = *(const v8bf*)&lds[p][640 + r][kb];
      bfr[g][1].h[1] = *(const v8bf*)&lds[p][640 + r][kb + 16];
    }

    // ---- per-mt A fragments + 12 WMMAs (3-pass split) ----
    #pragma unroll
    for (int mt = 0; mt < 4; ++mt) {
      int r  = wave_m * 64 + mt * 16 + l16;
      int kc = lhi * 8;
      FragU ah, al;
      ah.h[0] = *(const v8bf*)&lds[p][r][kc];
      ah.h[1] = *(const v8bf*)&lds[p][r][kc + 16];
      al.h[0] = *(const v8bf*)&lds[p][256 + r][kc];
      al.h[1] = *(const v8bf*)&lds[p][256 + r][kc + 16];
      #pragma unroll
      for (int g = 0; g < 4; ++g) {
        acc[mt][g] = __builtin_amdgcn_wmma_f32_16x16x32_bf16(
            false, ah.v, false, bfr[g][0].v, (short)0, acc[mt][g], false, false);
        acc[mt][g] = __builtin_amdgcn_wmma_f32_16x16x32_bf16(
            false, ah.v, false, bfr[g][1].v, (short)0, acc[mt][g], false, false);
        acc[mt][g] = __builtin_amdgcn_wmma_f32_16x16x32_bf16(
            false, al.v, false, bfr[g][0].v, (short)0, acc[mt][g], false, false);
      }
    }

    asm volatile("s_wait_asynccnt 0x0" ::: "memory");
    __syncthreads();
  }
  #undef ISSUE_TILE

  // ---- epilogue ----
  const int hn = h0 + wave_n * 16 + l16;
  float biasv[4];
  #pragma unroll
  for (int g = 0; g < 4; ++g)
    biasv[g] = bx[g * 1024 + hn] + bh[g * 1024 + hn];

  float* out_h = out;
  float* out_c = out + (size_t)BQ * HQ;

  #pragma unroll
  for (int mt = 0; mt < 4; ++mt) {
    const int mbase = row0 + wave_m * 64 + mt * 16 + lhi * 8;
    #pragma unroll
    for (int r = 0; r < 8; ++r) {
      const int b = mbase + r;
      float iv = sigmoidf_fast(acc[mt][0][r] + biasv[0]);
      float fv = sigmoidf_fast(acc[mt][1][r] + biasv[1]);
      float gv = tanhf_fast  (acc[mt][2][r] + biasv[2]);
      float ov = sigmoidf_fast(acc[mt][3][r] + biasv[3]);
      float cp = c_prev[(size_t)b * HQ + hn];
      float nc = fv * cp + iv * gv;
      float nh = ov * tanhf_fast(nc);
      out_h[(size_t)b * HQ + hn] = nh;
      out_c[(size_t)b * HQ + hn] = nc;
    }
  }
}

// ---------------------------------------------------------------------------
// Fallback (workspace too small): fused convert+GEMM kernel (2x4 blocking).
// ---------------------------------------------------------------------------
__global__ void __launch_bounds__(256)
lstm_cell_fused(const float* __restrict__ incoming,
                const float* __restrict__ state,
                const float* __restrict__ Wx,
                const float* __restrict__ bx,
                const float* __restrict__ Wh,
                const float* __restrict__ bh,
                float* __restrict__ out)
{
  __shared__ __bf16 Ahi[BMF][LDSS];
  __shared__ __bf16 Alo[BMF][LDSS];
  __shared__ __bf16 Bhi[4 * BH][LDSS];
  __shared__ __bf16 Blo[4 * BH][LDSS];

  const int tid    = threadIdx.x;
  const int lane   = tid & 31;
  const int wave   = tid >> 5;
  const int wave_m = wave >> 1;
  const int wave_n = wave & 1;
  const int l16    = lane & 15;
  const int lhi    = lane >> 4;

  const int row0 = blockIdx.x * BMF;
  const int h0   = blockIdx.y * BH;

  const float* h_prev = state;
  const float* c_prev = state + (size_t)BQ * HQ;

  v8f acc[2][4];
  #pragma unroll
  for (int mt = 0; mt < 2; ++mt)
    #pragma unroll
    for (int g = 0; g < 4; ++g)
      acc[mt][g] = (v8f){0.f,0.f,0.f,0.f,0.f,0.f,0.f,0.f};

  const float* Asrc[2] = { incoming, h_prev };
  const float* Wsrc[2] = { Wx, Wh };

  for (int phase = 0; phase < 2; ++phase) {
    const float* Ap = Asrc[phase];
    const float* Wp = Wsrc[phase];
    for (int kk = 0; kk < 1024; kk += KT) {
      __syncthreads();
      #pragma unroll
      for (int i = 0; i < 4; ++i) {
        int idx = tid + i * 256;
        int r = idx >> 3, c = (idx & 7) << 2;
        const float4 a = *(const float4*)(Ap + (size_t)(row0 + r) * 1024 + kk + c);
        float v[4] = { a.x, a.y, a.z, a.w };
        unsigned long long hp = 0ull, lp = 0ull;
        #pragma unroll
        for (int j = 0; j < 4; ++j) {
          unsigned short hb = f2bf_rn(v[j]);
          unsigned short lb = f2bf_rn(v[j] - bfbits2f(hb));
          hp |= (unsigned long long)hb << (16 * j);
          lp |= (unsigned long long)lb << (16 * j);
        }
        *(unsigned long long*)&Ahi[r][c] = hp;
        *(unsigned long long*)&Alo[r][c] = lp;
      }
      #pragma unroll
      for (int i = 0; i < 4; ++i) {
        int idx = tid + i * 256;
        int r = idx >> 3, gate = r >> 5, hl = r & 31, c = (idx & 7) << 2;
        const float4 w = *(const float4*)(Wp + (size_t)(gate * 1024 + h0 + hl) * 1024 + kk + c);
        float v[4] = { w.x, w.y, w.z, w.w };
        unsigned long long hp = 0ull, lp = 0ull;
        #pragma unroll
        for (int j = 0; j < 4; ++j) {
          unsigned short hb = f2bf_rn(v[j]);
          unsigned short lb = f2bf_rn(v[j] - bfbits2f(hb));
          hp |= (unsigned long long)hb << (16 * j);
          lp |= (unsigned long long)lb << (16 * j);
        }
        *(unsigned long long*)&Bhi[r][c] = hp;
        *(unsigned long long*)&Blo[r][c] = lp;
      }
      __syncthreads();

      FragU afr[2][2];
      #pragma unroll
      for (int mt = 0; mt < 2; ++mt) {
        int r = wave_m * 32 + mt * 16 + l16, kc = lhi * 8;
        afr[mt][0].h[0] = *(const v8bf*)&Ahi[r][kc];
        afr[mt][0].h[1] = *(const v8bf*)&Ahi[r][kc + 16];
        afr[mt][1].h[0] = *(const v8bf*)&Alo[r][kc];
        afr[mt][1].h[1] = *(const v8bf*)&Alo[r][kc + 16];
      }
      FragU bfr[4][2];
      #pragma unroll
      for (int g = 0; g < 4; ++g) {
        int r = g * 32 + wave_n * 16 + l16, kb = lhi * 8;
        bfr[g][0].h[0] = *(const v8bf*)&Bhi[r][kb];
        bfr[g][0].h[1] = *(const v8bf*)&Bhi[r][kb + 16];
        bfr[g][1].h[0] = *(const v8bf*)&Blo[r][kb];
        bfr[g][1].h[1] = *(const v8bf*)&Blo[r][kb + 16];
      }
      #pragma unroll
      for (int mt = 0; mt < 2; ++mt)
        #pragma unroll
        for (int g = 0; g < 4; ++g) {
          acc[mt][g] = __builtin_amdgcn_wmma_f32_16x16x32_bf16(
              false, afr[mt][0].v, false, bfr[g][0].v, (short)0, acc[mt][g], false, false);
          acc[mt][g] = __builtin_amdgcn_wmma_f32_16x16x32_bf16(
              false, afr[mt][0].v, false, bfr[g][1].v, (short)0, acc[mt][g], false, false);
          acc[mt][g] = __builtin_amdgcn_wmma_f32_16x16x32_bf16(
              false, afr[mt][1].v, false, bfr[g][0].v, (short)0, acc[mt][g], false, false);
        }
    }
  }

  const int hn = h0 + wave_n * 16 + l16;
  float biasv[4];
  #pragma unroll
  for (int g = 0; g < 4; ++g)
    biasv[g] = bx[g * 1024 + hn] + bh[g * 1024 + hn];

  float* out_h = out;
  float* out_c = out + (size_t)BQ * HQ;

  #pragma unroll
  for (int mt = 0; mt < 2; ++mt) {
    const int mbase = row0 + wave_m * 32 + mt * 16 + lhi * 8;
    #pragma unroll
    for (int r = 0; r < 8; ++r) {
      const int b = mbase + r;
      float iv = sigmoidf_fast(acc[mt][0][r] + biasv[0]);
      float fv = sigmoidf_fast(acc[mt][1][r] + biasv[1]);
      float gv = tanhf_fast  (acc[mt][2][r] + biasv[2]);
      float ov = sigmoidf_fast(acc[mt][3][r] + biasv[3]);
      float cp = c_prev[(size_t)b * HQ + hn];
      float nc = fv * cp + iv * gv;
      float nh = ov * tanhf_fast(nc);
      out_h[(size_t)b * HQ + hn] = nh;
      out_c[(size_t)b * HQ + hn] = nc;
    }
  }
}

extern "C" void kernel_launch(void* const* d_in, const int* in_sizes, int n_in,
                              void* d_out, int out_size, void* d_ws, size_t ws_size,
                              hipStream_t stream) {
  const float* incoming = (const float*)d_in[0];
  const float* state    = (const float*)d_in[1];
  const float* Wx       = (const float*)d_in[2];
  const float* bx       = (const float*)d_in[3];
  const float* Wh       = (const float*)d_in[4];
  const float* bh       = (const float*)d_in[5];
  float* out = (float*)d_out;

  const size_t nX = (size_t)BQ * KQ;           // 16M elems
  const size_t nW = (size_t)(4 * HQ) * KQ;     // 8M elems
  const size_t ws_need = (2 * nX + 2 * nW) * sizeof(__bf16);  // 96 MiB

  if (ws_size >= ws_need) {
    __bf16* Xhi = (__bf16*)d_ws;
    __bf16* Xlo = Xhi + nX;
    __bf16* Whi = Xlo + nX;
    __bf16* Wlo = Whi + nW;
    // Pass 1: split-convert X = [incoming | h_prev], W = [Wx | Wh]
    hipLaunchKernelGGL(split_bf16_kernel, dim3((unsigned)(nX / 4 / 256)), dim3(256),
                       0, stream, incoming, state /*h_prev*/, Xhi, Xlo, BQ);
    hipLaunchKernelGGL(split_bf16_kernel, dim3((unsigned)(nW / 4 / 256)), dim3(256),
                       0, stream, Wx, Wh, Whi, Wlo, 4 * HQ);
    // Pass 2: GEMM + epilogue (4x4 wave blocking, async double buffer)
    hipLaunchKernelGGL(lstm_gemm_bf16, dim3(BQ / BMG, HQ / BH), dim3(256),
                       0, stream, Xhi, state, bx, bh, out);
  } else {
    hipLaunchKernelGGL(lstm_cell_fused, dim3(BQ / BMF, HQ / BH), dim3(256),
                       0, stream, incoming, state, Wx, bx, Wh, bh, out);
  }
}
